// Decoder_38809324487444
// MI455X (gfx1250) — compile-verified, matrix operand order checked by reading, not run on hardware
//
#include <hip/hip_runtime.h>
#include <hip/hip_bf16.h>

// ---------------- problem constants ----------------
#define NCITY  50
#define BSZ    1024
#define HDIM   256
#define EDIM   256
#define SFDIM  5
#define STEPS  50
#define CTANH  10.0f

typedef __attribute__((ext_vector_type(16))) __bf16        v16bf;
typedef __attribute__((ext_vector_type(8)))  float         v8f;
typedef __attribute__((ext_vector_type(4)))  unsigned int  u32x4;
typedef __attribute__((ext_vector_type(2)))  unsigned int  u32x2;

// ---------------- bf16 helpers (bit-level, RNE) ----------------
__device__ __forceinline__ unsigned short f2b(float f) {
    union { float f; unsigned int u; } x; x.f = f;
    unsigned int u = x.u;
    unsigned int r = (u + 0x7FFFu + ((u >> 16) & 1u)) >> 16;
    return (unsigned short)r;
}
__device__ __forceinline__ float b2f(unsigned short b) {
    union { unsigned int u; float f; } x; x.u = ((unsigned int)b) << 16;
    return x.f;
}
// unpack low/high bf16 of a packed u32
__device__ __forceinline__ float blo(unsigned int u) {
    union { unsigned int u; float f; } x; x.u = u << 16; return x.f;
}
__device__ __forceinline__ float bhi(unsigned int u) {
    union { unsigned int u; float f; } x; x.u = u & 0xffff0000u; return x.f;
}
__device__ __forceinline__ unsigned int pk2(float a, float b) {
    return (unsigned int)f2b(a) | ((unsigned int)f2b(b) << 16);
}
__device__ __forceinline__ float sigm(float x) { return 1.0f / (1.0f + __expf(-x)); }

// ---------------- WMMA helpers ----------------
__device__ __forceinline__ v8f wmma_bf16(v8f acc, v16bf a, v16bf b) {
    return __builtin_amdgcn_wmma_f32_16x16x32_bf16(false, a, false, b,
                                                   (short)0, acc, false, false);
}
__device__ __forceinline__ v16bf ldA(const unsigned short* s, int lane, int kc) {
    const unsigned short* p = s + (lane & 15) * HDIM + kc * 32 + ((lane >> 4) << 4);
    return *(const v16bf*)p;
}
__device__ __forceinline__ v16bf ldB(const unsigned short* w, int lane, int jtg, int kc) {
    const unsigned short* p = w + ((size_t)(jtg * 8 + kc)) * 512 + lane * 16;
    return *(const v16bf*)p;
}
__device__ __forceinline__ void store_tile(unsigned short* sOut, int jt, int lane,
                                           const float h[8]) {
    int j = jt * 16 + (lane & 15);
    int hi = (lane >> 4) << 3;
#pragma unroll
    for (int r = 0; r < 8; ++r) sOut[(r + hi) * HDIM + j] = f2b(h[r]);
}

// One LSTM cell column-tile (16 hidden units) for a 16-row batch tile.
__device__ __forceinline__ void lstm_gates(
    const unsigned short* sIn, const unsigned short* sHid,
    const unsigned short* wI, const unsigned short* wH,
    const float* bI, const float* bH,
    int jt, int lane, v8f& c, float hOut[8]) {
    int j = jt * 16 + (lane & 15);
    v8f g[4];
#pragma unroll
    for (int gg = 0; gg < 4; ++gg) {
        float bias = bI[gg * HDIM + j] + bH[gg * HDIM + j];
        v8f acc;
#pragma unroll
        for (int r = 0; r < 8; ++r) acc[r] = bias;
        int jtg = gg * 16 + jt;
#pragma unroll
        for (int kc = 0; kc < 8; ++kc)
            acc = wmma_bf16(acc, ldA(sIn, lane, kc), ldB(wI, lane, jtg, kc));
#pragma unroll
        for (int kc = 0; kc < 8; ++kc)
            acc = wmma_bf16(acc, ldA(sHid, lane, kc), ldB(wH, lane, jtg, kc));
        g[gg] = acc;
    }
#pragma unroll
    for (int r = 0; r < 8; ++r) {
        float iv = sigm(g[0][r]);
        float fv = sigm(g[1][r]);
        float gv = tanhf(g[2][r]);
        float ov = sigm(g[3][r]);
        float c2 = fv * c[r] + iv * gv;
        c[r]     = c2;
        hOut[r]  = ov * tanhf(c2);
    }
}

__device__ __forceinline__ void proj_store(
    const unsigned short* sIn, const unsigned short* wQ, const float* bQ,
    unsigned short* sOut, int jt, int lane) {
    int j = jt * 16 + (lane & 15);
    float bias = bQ[j];
    v8f acc;
#pragma unroll
    for (int r = 0; r < 8; ++r) acc[r] = bias;
#pragma unroll
    for (int kc = 0; kc < 8; ++kc)
        acc = wmma_bf16(acc, ldA(sIn, lane, kc), ldB(wQ, lane, jt, kc));
    float h[8];
#pragma unroll
    for (int r = 0; r < 8; ++r) h[r] = acc[r];
    store_tile(sOut, jt, lane, h);
}

// vectorized attention dot:  sum_h v[h] * tanh(q[row,h] + e[h])   (128-bit loads)
__device__ __forceinline__ float attn_dot(const unsigned short* qrow,
                                          const unsigned short* e,
                                          const float* __restrict__ v) {
    float acc = 0.f;
#pragma unroll 4
    for (int h = 0; h < HDIM; h += 8) {
        u32x4 eq = *(const u32x4*)(e + h);      // global_load_b128 (8 bf16)
        u32x4 qq = *(const u32x4*)(qrow + h);   // ds_load_b128     (8 bf16)
#pragma unroll
        for (int i = 0; i < 4; ++i) {
            float v0 = v[h + 2 * i], v1 = v[h + 2 * i + 1];
            acc += v0 * tanhf(blo(qq[i]) + blo(eq[i]));
            acc += v1 * tanhf(bhi(qq[i]) + bhi(eq[i]));
        }
    }
    return acc;
}

// ---------------- weight swizzle: f32 [outdim,K] -> bf16 WMMA-B blocks ----------------
__global__ void swz_kernel(const float* __restrict__ W, unsigned short* __restrict__ out,
                           int outdim, int K) {
    int e = blockIdx.x * blockDim.x + threadIdx.x;
    int total = outdim * K;
    if (e >= total) return;
    int blk = e >> 9;
    int inb = e & 511;
    int l = inb >> 4, v = inb & 15;
    int kT = K >> 5;
    int jt = blk / kT, kc = blk - jt * kT;
    int j  = jt * 16 + (l & 15);
    int kk = kc * 32 + ((l >> 4) << 4) + v;
    out[e] = f2b(W[(size_t)j * K + kk]);
}

// ---------------- precompute e_ptr / e_glp: context @ Wr.T + br, bf16 [b][n][h] ------
__global__ __launch_bounds__(256) void eprep_kernel(
    const float* __restrict__ context,
    const unsigned short* __restrict__ w_ptr_wr, const float* __restrict__ ptr_br,
    const unsigned short* __restrict__ w_glp_wr, const float* __restrict__ glp_br,
    unsigned short* __restrict__ e_ptr, unsigned short* __restrict__ e_glp) {
    __shared__ __align__(32) unsigned short sCtx[16 * HDIM];
    int tid = threadIdx.x, lane = tid & 31, wave = tid >> 5;
    int rbase = blockIdx.x * 16;                       // row r = b*NCITY + n
    for (int t = tid; t < 16 * 64; t += 256) {
        int rowl = t >> 6, hb = (t & 63) * 4;
        int r = rbase + rowl;
        int b = r / NCITY, n = r - b * NCITY;
        const float4 cv = *(const float4*)(context + ((size_t)n * BSZ + b) * HDIM + hb);
        *(u32x2*)(sCtx + rowl * HDIM + hb) = (u32x2){pk2(cv.x, cv.y), pk2(cv.z, cv.w)};
    }
    __syncthreads();
#pragma unroll
    for (int jj = 0; jj < 2; ++jj) {
        int jt = wave * 2 + jj;
        int j  = jt * 16 + (lane & 15);
        float bp = ptr_br[j], bg = glp_br[j];
        v8f aP, aG;
#pragma unroll
        for (int r = 0; r < 8; ++r) { aP[r] = bp; aG[r] = bg; }
#pragma unroll
        for (int kc = 0; kc < 8; ++kc) {
            v16bf a = ldA(sCtx, lane, kc);
            aP = wmma_bf16(aP, a, ldB(w_ptr_wr, lane, jt, kc));
            aG = wmma_bf16(aG, a, ldB(w_glp_wr, lane, jt, kc));
        }
        int hi = (lane >> 4) << 3;
#pragma unroll
        for (int r = 0; r < 8; ++r) {
            size_t rg = (size_t)(rbase + r + hi);
            e_ptr[rg * HDIM + j] = f2b(aP[r]);
            e_glp[rg * HDIM + j] = f2b(aG[r]);
        }
    }
}

// ---------------- fused persistent decoder: 16 batch rows per block, 50 steps -------
__global__ __launch_bounds__(256) void decoder_kernel(
    const float* __restrict__ start_fea, const float* __restrict__ decoder_input,
    const float* __restrict__ embedded,  const float* __restrict__ hidden_h,
    const float* __restrict__ hidden_c,  const float* __restrict__ context,
    const unsigned char* __restrict__ vmask,
    const float* __restrict__ lstm_b_ih, const float* __restrict__ lstm_b_hh,
    const float* __restrict__ ptr_bq,    const float* __restrict__ ptr_v,
    const float* __restrict__ glp_bq,    const float* __restrict__ glp_v,
    const float* __restrict__ fn_w,
    const float* __restrict__ eta_b_ih,  const float* __restrict__ eta_b_hh,
    const float* __restrict__ eta_lin_w, const float* __restrict__ eta_lin_b,
    const unsigned short* __restrict__ w_lstm_ih, const unsigned short* __restrict__ w_lstm_hh,
    const unsigned short* __restrict__ w_eta_ih,  const unsigned short* __restrict__ w_eta_hh,
    const unsigned short* __restrict__ w_ptr_wq,  const unsigned short* __restrict__ w_glp_wq,
    const unsigned short* __restrict__ e_ptr,     const unsigned short* __restrict__ e_glp,
    float* __restrict__ outLogp, float* __restrict__ outSel, float* __restrict__ outEta) {

    __shared__ __align__(32) unsigned short sX[16 * HDIM];
    __shared__ __align__(32) unsigned short sH[16 * HDIM];
    __shared__ __align__(32) unsigned short sHe[16 * HDIM];
    __shared__ __align__(32) unsigned short sCur[16 * HDIM];
    __shared__ __align__(32) unsigned short sLast[16 * HDIM];
    __shared__ __align__(32) unsigned short sQ[16 * HDIM];
    __shared__ __align__(32) unsigned short sG[16 * HDIM];
    __shared__ float sU[16 * 64];
    __shared__ unsigned char sMask[16 * 64];
    __shared__ int   sIdx[16];
    __shared__ float sEtaAcc[16];

    int tid = threadIdx.x, lane = tid & 31, wave = tid >> 5;
    int bbase = blockIdx.x * 16;
    int jt0 = wave * 2, jt1 = wave * 2 + 1;

    // warm L2 for this block's attention tables (global_prefetch path)
    if (tid < NCITY) {
        __builtin_prefetch(e_glp + ((size_t)bbase * NCITY + tid) * HDIM, 0, 1);
        __builtin_prefetch(e_ptr + ((size_t)bbase * NCITY + tid) * HDIM, 0, 1);
    }

    // ---- init: x, h, he, first_node ----
    for (int t = tid; t < 16 * HDIM; t += 256) {
        int row = t >> 8, h = t & 255;
        int gb = bbase + row;
        sX[t] = f2b(decoder_input[(size_t)gb * EDIM + h]);
        unsigned short hb = f2b(hidden_h[(size_t)gb * HDIM + h]);
        sH[t] = hb; sHe[t] = hb;
        float fn = 0.f;
#pragma unroll
        for (int k = 0; k < SFDIM; ++k)
            fn += start_fea[gb * SFDIM + k] * fn_w[h * SFDIM + k];
        unsigned short fnb = f2b(fn);
        sCur[t] = fnb; sLast[t] = fnb;
    }
    if (tid < 16) {
        int gb = bbase + tid;
        int cnt = 0;
        for (int n = 0; n < NCITY; ++n) {
            unsigned char m = vmask[(size_t)gb * NCITY + n] ? 1 : 0;
            sMask[tid * 64 + n] = m; cnt += m;
        }
        for (int n = NCITY; n < 64; ++n) sMask[tid * 64 + n] = 1;
        if (cnt == NCITY) sMask[tid * 64 + NCITY - 1] = 0;   // _mask_modify
        sIdx[tid] = 0;
        sEtaAcc[tid] = 0.f;
    }
    // ---- cell states live in registers in WMMA-D layout ----
    v8f cP0, cP1, cE0, cE1;
    {
        int hi = (lane >> 4) << 3;
        int j0 = jt0 * 16 + (lane & 15), j1 = jt1 * 16 + (lane & 15);
#pragma unroll
        for (int r = 0; r < 8; ++r) {
            int row = r + hi;
            float c0 = hidden_c[(size_t)(bbase + row) * HDIM + j0];
            float c1 = hidden_c[(size_t)(bbase + row) * HDIM + j1];
            cP0[r] = c0; cP1[r] = c1; cE0[r] = c0; cE1[r] = c1;
        }
    }
    __syncthreads();

    // ---- initial eta-LSTM step on first_node ----
    {
        float h0[8], h1[8];
        lstm_gates(sCur, sHe, w_eta_ih, w_eta_hh, eta_b_ih, eta_b_hh, jt0, lane, cE0, h0);
        lstm_gates(sCur, sHe, w_eta_ih, w_eta_hh, eta_b_ih, eta_b_hh, jt1, lane, cE1, h1);
        __syncthreads();
        store_tile(sHe, jt0, lane, h0);
        store_tile(sHe, jt1, lane, h1);
        __syncthreads();
    }

    const float NEG_INF = -__builtin_inff();

    for (int s = 0; s < STEPS; ++s) {
        // (a) mask update (skip at step 0)
        if (s > 0 && tid < 16) {
            sMask[tid * 64 + sIdx[tid]] = 1;
            int cnt = 0;
            for (int n = 0; n < NCITY; ++n) cnt += sMask[tid * 64 + n];
            if (cnt == NCITY) sMask[tid * 64 + NCITY - 1] = 0;
        }
        // (b) pointer LSTM
        float h0[8], h1[8];
        lstm_gates(sX, sH, w_lstm_ih, w_lstm_hh, lstm_b_ih, lstm_b_hh, jt0, lane, cP0, h0);
        lstm_gates(sX, sH, w_lstm_ih, w_lstm_hh, lstm_b_ih, lstm_b_hh, jt1, lane, cP1, h1);
        __syncthreads();
        store_tile(sH, jt0, lane, h0);
        store_tile(sH, jt1, lane, h1);
        __syncthreads();
        // (c) glimpse query projection
        proj_store(sH, w_glp_wq, glp_bq, sQ, jt0, lane);
        proj_store(sH, w_glp_wq, glp_bq, sQ, jt1, lane);
        __syncthreads();
        // (d) glimpse logits (vectorized 128-bit loads)
        for (int p = tid; p < 16 * NCITY; p += 256) {
            int row = p / NCITY, n = p - row * NCITY;
            const unsigned short* e = e_glp + (((size_t)(bbase + row)) * NCITY + n) * HDIM;
            float acc = attn_dot(sQ + row * HDIM, e, glp_v);
            sU[row * 64 + n] = sMask[row * 64 + n] ? NEG_INF : acc;
        }
        __syncthreads();
        // (e) softmax over n (in place -> probs)
        if (tid < 16) {
            float m = NEG_INF;
            for (int n = 0; n < NCITY; ++n) m = fmaxf(m, sU[tid * 64 + n]);
            float ssum = 0.f;
            for (int n = 0; n < NCITY; ++n) {
                float pz = __expf(sU[tid * 64 + n] - m);
                sU[tid * 64 + n] = pz; ssum += pz;
            }
            float inv = 1.f / ssum;
            for (int n = 0; n < NCITY; ++n) sU[tid * 64 + n] *= inv;
        }
        __syncthreads();
        // (f) g_l[b,h] = sum_n p[b,n] * e_glp[b,n,h]  (8 h per task, b128 loads)
        for (int t = tid; t < 16 * 32; t += 256) {
            int row = t >> 5, hb = (t & 31) * 8;
            int gb = bbase + row;
            const unsigned short* e = e_glp + ((size_t)gb * NCITY) * HDIM + hb;
            float acc[8];
#pragma unroll
            for (int i = 0; i < 8; ++i) acc[i] = 0.f;
            for (int n = 0; n < NCITY; ++n) {
                u32x4 ev = *(const u32x4*)(e + (size_t)n * HDIM);
                float p = sU[row * 64 + n];
#pragma unroll
                for (int i = 0; i < 4; ++i) {
                    acc[2 * i]     += p * blo(ev[i]);
                    acc[2 * i + 1] += p * bhi(ev[i]);
                }
            }
            u32x4 outv;
#pragma unroll
            for (int i = 0; i < 4; ++i) outv[i] = pk2(acc[2 * i], acc[2 * i + 1]);
            *(u32x4*)(sG + row * HDIM + hb) = outv;
        }
        __syncthreads();
        // (g) pointer query projection
        proj_store(sG, w_ptr_wq, ptr_bq, sQ, jt0, lane);
        proj_store(sG, w_ptr_wq, ptr_bq, sQ, jt1, lane);
        __syncthreads();
        // (h) pointer logits: C * tanh(u), masked
        for (int p = tid; p < 16 * NCITY; p += 256) {
            int row = p / NCITY, n = p - row * NCITY;
            const unsigned short* e = e_ptr + (((size_t)(bbase + row)) * NCITY + n) * HDIM;
            float acc = attn_dot(sQ + row * HDIM, e, ptr_v);
            sU[row * 64 + n] = sMask[row * 64 + n] ? NEG_INF : CTANH * tanhf(acc);
        }
        __syncthreads();
        // (i) log_softmax + argmax + outputs
        if (tid < 16) {
            int gb = bbase + tid;
            float m = NEG_INF; int am = 0;
            for (int n = 0; n < NCITY; ++n) {
                float u = sU[tid * 64 + n];
                if (u > m) { m = u; am = n; }
            }
            float ssum = 0.f;
            for (int n = 0; n < NCITY; ++n) ssum += __expf(sU[tid * 64 + n] - m);
            float lse = __logf(ssum);
            for (int n = 0; n < NCITY; ++n)
                outLogp[((size_t)gb * STEPS + s) * NCITY + n] = sU[tid * 64 + n] - m - lse;
            sIdx[tid] = am;
            outSel[(size_t)gb * STEPS + s] = (float)am;
        }
        __syncthreads();
        // (j) gather cur_node & next decoder input (float4 loads, packed bf16 stores)
        for (int t = tid; t < 16 * 64; t += 256) {
            int row = t >> 6, hb = (t & 63) * 4;
            int gb = bbase + row;
            int idx = sIdx[row];
            const float4 cv = *(const float4*)(context  + ((size_t)idx * BSZ + gb) * HDIM + hb);
            const float4 xv = *(const float4*)(embedded + ((size_t)idx * BSZ + gb) * EDIM + hb);
            *(u32x2*)(sCur + row * HDIM + hb) = (u32x2){pk2(cv.x, cv.y), pk2(cv.z, cv.w)};
            *(u32x2*)(sX   + row * HDIM + hb) = (u32x2){pk2(xv.x, xv.y), pk2(xv.z, xv.w)};
        }
        __syncthreads();
        // (k) eta linear + accumulate (vectorized LDS reads)
        if (tid < 16) {
            int gb = bbase + tid;
            float acc = eta_lin_b[0];
#pragma unroll 4
            for (int h = 0; h < HDIM; h += 8) {
                u32x4 lv = *(const u32x4*)(sLast + tid * HDIM + h);
                u32x4 cv = *(const u32x4*)(sCur  + tid * HDIM + h);
#pragma unroll
                for (int i = 0; i < 4; ++i) {
                    acc += blo(lv[i]) * eta_lin_w[h + 2 * i]
                         + bhi(lv[i]) * eta_lin_w[h + 2 * i + 1];
                    acc += blo(cv[i]) * eta_lin_w[HDIM + h + 2 * i]
                         + bhi(cv[i]) * eta_lin_w[HDIM + h + 2 * i + 1];
                }
            }
            sEtaAcc[tid] += acc;
            outEta[(size_t)gb * STEPS + s] = sEtaAcc[tid];
        }
        __syncthreads();
        // (l) eta LSTM + last_node <- cur_node
        lstm_gates(sCur, sHe, w_eta_ih, w_eta_hh, eta_b_ih, eta_b_hh, jt0, lane, cE0, h0);
        lstm_gates(sCur, sHe, w_eta_ih, w_eta_hh, eta_b_ih, eta_b_hh, jt1, lane, cE1, h1);
        for (int t = tid; t < 16 * 32; t += 256) {
            int row = t >> 5, hb = (t & 31) * 8;
            *(u32x4*)(sLast + row * HDIM + hb) = *(const u32x4*)(sCur + row * HDIM + hb);
        }
        __syncthreads();
        store_tile(sHe, jt0, lane, h0);
        store_tile(sHe, jt1, lane, h1);
        __syncthreads();
    }
}

// ---------------- workspace layout (bytes) ----------------
#define OFF_LSTM_IH 0u
#define OFF_LSTM_HH (OFF_LSTM_IH + 1024u*256u*2u)
#define OFF_ETA_IH  (OFF_LSTM_HH + 1024u*256u*2u)
#define OFF_ETA_HH  (OFF_ETA_IH  + 1024u*256u*2u)
#define OFF_PTR_WQ  (OFF_ETA_HH  + 1024u*256u*2u)
#define OFF_GLP_WQ  (OFF_PTR_WQ  + 256u*256u*2u)
#define OFF_PTR_WR  (OFF_GLP_WQ  + 256u*256u*2u)
#define OFF_GLP_WR  (OFF_PTR_WR  + 256u*256u*2u)
#define OFF_EPTR    (OFF_GLP_WR  + 256u*256u*2u)
#define OFF_EGLP    (OFF_EPTR    + (size_t)BSZ*NCITY*HDIM*2u)

extern "C" void kernel_launch(void* const* d_in, const int* in_sizes, int n_in,
                              void* d_out, int out_size, void* d_ws, size_t ws_size,
                              hipStream_t stream) {
    const float* start_fea     = (const float*)d_in[0];
    const float* decoder_input = (const float*)d_in[1];
    const float* embedded      = (const float*)d_in[2];
    const float* hidden_h      = (const float*)d_in[3];
    const float* hidden_c      = (const float*)d_in[4];
    const float* context       = (const float*)d_in[5];
    // d_in[6] = V (unused by the reference)
    const unsigned char* vmask = (const unsigned char*)d_in[7];
    const float* lstm_w_ih = (const float*)d_in[8];
    const float* lstm_w_hh = (const float*)d_in[9];
    const float* lstm_b_ih = (const float*)d_in[10];
    const float* lstm_b_hh = (const float*)d_in[11];
    const float* ptr_wq = (const float*)d_in[12];
    const float* ptr_bq = (const float*)d_in[13];
    const float* ptr_wr = (const float*)d_in[14];
    const float* ptr_br = (const float*)d_in[15];
    const float* ptr_v  = (const float*)d_in[16];
    const float* glp_wq = (const float*)d_in[17];
    const float* glp_bq = (const float*)d_in[18];
    const float* glp_wr = (const float*)d_in[19];
    const float* glp_br = (const float*)d_in[20];
    const float* glp_v  = (const float*)d_in[21];
    const float* fn_w   = (const float*)d_in[22];
    const float* eta_w_ih = (const float*)d_in[23];
    const float* eta_w_hh = (const float*)d_in[24];
    const float* eta_b_ih = (const float*)d_in[25];
    const float* eta_b_hh = (const float*)d_in[26];
    const float* eta_lin_w = (const float*)d_in[27];
    const float* eta_lin_b = (const float*)d_in[28];

    unsigned char* ws = (unsigned char*)d_ws;
    unsigned short* w_lstm_ih = (unsigned short*)(ws + OFF_LSTM_IH);
    unsigned short* w_lstm_hh = (unsigned short*)(ws + OFF_LSTM_HH);
    unsigned short* w_eta_ih  = (unsigned short*)(ws + OFF_ETA_IH);
    unsigned short* w_eta_hh  = (unsigned short*)(ws + OFF_ETA_HH);
    unsigned short* w_ptr_wq  = (unsigned short*)(ws + OFF_PTR_WQ);
    unsigned short* w_glp_wq  = (unsigned short*)(ws + OFF_GLP_WQ);
    unsigned short* w_ptr_wr  = (unsigned short*)(ws + OFF_PTR_WR);
    unsigned short* w_glp_wr  = (unsigned short*)(ws + OFF_GLP_WR);
    unsigned short* e_ptr     = (unsigned short*)(ws + OFF_EPTR);
    unsigned short* e_glp     = (unsigned short*)(ws + OFF_EGLP);

    float* outLogp = (float*)d_out;
    float* outSel  = outLogp + (size_t)BSZ * STEPS * NCITY;
    float* outEta  = outSel  + (size_t)BSZ * STEPS;

    int big = 1024 * 256, small = 256 * 256;
    swz_kernel<<<(big + 255) / 256, 256, 0, stream>>>(lstm_w_ih, w_lstm_ih, 1024, 256);
    swz_kernel<<<(big + 255) / 256, 256, 0, stream>>>(lstm_w_hh, w_lstm_hh, 1024, 256);
    swz_kernel<<<(big + 255) / 256, 256, 0, stream>>>(eta_w_ih,  w_eta_ih,  1024, 256);
    swz_kernel<<<(big + 255) / 256, 256, 0, stream>>>(eta_w_hh,  w_eta_hh,  1024, 256);
    swz_kernel<<<(small + 255) / 256, 256, 0, stream>>>(ptr_wq, w_ptr_wq, 256, 256);
    swz_kernel<<<(small + 255) / 256, 256, 0, stream>>>(glp_wq, w_glp_wq, 256, 256);
    swz_kernel<<<(small + 255) / 256, 256, 0, stream>>>(ptr_wr, w_ptr_wr, 256, 256);
    swz_kernel<<<(small + 255) / 256, 256, 0, stream>>>(glp_wr, w_glp_wr, 256, 256);

    eprep_kernel<<<(BSZ * NCITY) / 16, 256, 0, stream>>>(
        context, w_ptr_wr, ptr_br, w_glp_wr, glp_br, e_ptr, e_glp);

    decoder_kernel<<<BSZ / 16, 256, 0, stream>>>(
        start_fea, decoder_input, embedded, hidden_h, hidden_c, context, vmask,
        lstm_b_ih, lstm_b_hh, ptr_bq, ptr_v, glp_bq, glp_v, fn_w,
        eta_b_ih, eta_b_hh, eta_lin_w, eta_lin_b,
        w_lstm_ih, w_lstm_hh, w_eta_ih, w_eta_hh, w_ptr_wq, w_glp_wq,
        e_ptr, e_glp, outLogp, outSel, outEta);
}
